// Attn_55516747268530
// MI455X (gfx1250) — matrix operand order repaired and checked by reading, not compiled, not run on hardware
//
#include <hip/hip_runtime.h>
#include <hip/hip_bf16.h>
#include <math.h>

#define NPTS 32768
#define NHEAD 8
#define DH 64
#define NBUCK 256          // NPTS / 128
#define LHN 16             // L * H

typedef _Float16 v16h __attribute__((ext_vector_type(16)));
typedef _Float16 v8h  __attribute__((ext_vector_type(8)));
typedef float    v8f  __attribute__((ext_vector_type(8)));

// ---------------- WMMA fragment loaders (CDNA5 16x16x32 f16 layouts) ----------

// A: 16x32 (MxK) row-major, lda elements. lane<16: M=lane, halves0-15=K0-15;
// lane>=16: M=lane-16, halves0-15=K16-31.
__device__ __forceinline__ v16h load_a_frag(const _Float16* A, int lda, int lane) {
  int row = lane & 15;
  int kb  = (lane & 16) ? 16 : 0;
  const _Float16* p = A + row * lda + kb;
  v8h lo = *(const v8h*)p;
  v8h hi = *(const v8h*)(p + 8);
  return __builtin_shufflevector(lo, hi, 0, 1, 2, 3, 4, 5, 6, 7,
                                 8, 9, 10, 11, 12, 13, 14, 15);
}

// B fragment from a row-major NxK buffer (B[k][n] = Bt[n*ldb + k]); 16
// contiguous halves per lane -> vector loads.
__device__ __forceinline__ v16h load_bT_frag(const _Float16* Bt, int ldb, int lane) {
  int col = lane & 15;
  int kb  = (lane & 16) ? 16 : 0;
  const _Float16* p = Bt + col * ldb + kb;
  v16h b;
#pragma unroll
  for (int i = 0; i < 16; ++i) b[i] = p[i];
  return b;
}

#define WMMA_F16(a, b, c) \
  __builtin_amdgcn_wmma_f32_16x16x32_f16(false, (a), false, (b), (short)0, (c), false, false)

// ---------------- CDNA5 async global->LDS DMA (ASYNCcnt-tracked) -------------

__device__ __forceinline__ void async_load_b128(unsigned lds_off, const void* gaddr) {
  asm volatile("global_load_async_to_lds_b128 %0, %1, off"
               :: "v"(lds_off), "v"((unsigned long long)(uintptr_t)gaddr)
               : "memory");
}
__device__ __forceinline__ void wait_asynccnt0() {
  asm volatile("s_wait_asynccnt 0x0" ::: "memory");
}

// ---------------- small utility kernels --------------------------------------

// Pack f32 weight W (KxNc row-major) into WMMA-B fragment order, f16:
// Wp[((kt*NT)+tn)*512 + lane*16 + i] = W[(kt*32 + kb(lane) + i)*Nc + tn*16 + (lane&15)]
__global__ void pack_weight_kernel(const float* __restrict__ W, _Float16* __restrict__ Wp,
                                   int K, int Nc) {
  int gid = blockIdx.x * blockDim.x + threadIdx.x;
  if (gid >= K * Nc) return;
  int NT = Nc >> 4;
  int i    = gid & 15;
  int lane = (gid >> 4) & 31;
  int tile = gid >> 9;
  int tn = tile % NT, kt = tile / NT;
  int col = lane & 15;
  int kb  = (lane & 16) ? 16 : 0;
  int k = kt * 32 + kb + i;
  int n = tn * 16 + col;
  Wp[gid] = (_Float16)W[(size_t)k * Nc + n];
}

// RPE scale: sqrt(2 * sum_k exp(min(sum_d w4[h,d,r,k], 50)))  -> 24 values
__global__ void rpe_kernel(const float* __restrict__ w_rpe, float* __restrict__ sqrtw) {
  int t = threadIdx.x;
  if (t >= NHEAD * 3) return;
  int h = t / 3, r = t % 3;
  float qw = 0.f;
  for (int k = 0; k < 8; ++k) {
    float s = 0.f;
    for (int d = 0; d < DH; ++d) s += w_rpe[(h * DH + d) * 24 + r * 8 + k];
    qw += __expf(fminf(s, 50.f));
  }
  sqrtw[t] = sqrtf(2.f * qw);
}

// LayerNorm (rows of 64), one wave per row, f32 in -> f16 out
__global__ void ln_kernel(const float* __restrict__ x, const float* __restrict__ g,
                          const float* __restrict__ b, _Float16* __restrict__ out, int nrows) {
  int wid  = (blockIdx.x * blockDim.x + threadIdx.x) >> 5;
  int lane = threadIdx.x & 31;
  if (wid >= nrows) return;
  const float* row = x + wid * DH;
  float a0 = row[lane], a1 = row[lane + 32];
  float s = a0 + a1;
#pragma unroll
  for (int m = 16; m; m >>= 1) s += __shfl_xor(s, m);
  float mu = s * (1.f / 64.f);
  float d0 = a0 - mu, d1 = a1 - mu;
  float v = d0 * d0 + d1 * d1;
#pragma unroll
  for (int m = 16; m; m >>= 1) v += __shfl_xor(v, m);
  float inv = rsqrtf(v * (1.f / 64.f) + 1e-5f);
  out[wid * DH + lane]      = (_Float16)(d0 * inv * g[lane] + b[lane]);
  out[wid * DH + lane + 32] = (_Float16)(d1 * inv * g[lane + 32] + b[lane + 32]);
}

// ---------------- generic WMMA GEMM (packed-B, software pipelined) ------------
// C(MxNc) = A(MxK,f16) @ Bp(packed fragments); mode 0: store f16
// mode 1: out_f = acc + bias[n] + resid[m*Nc+n]   mode 2: out_h = silu(acc + bias[n])
__global__ void gemm_wmma(const _Float16* __restrict__ A, const _Float16* __restrict__ Bp,
                          int M, int K, int Nc,
                          const float* __restrict__ bias, const float* __restrict__ resid,
                          _Float16* __restrict__ out_h, float* __restrict__ out_f, int mode) {
  int wid  = (blockIdx.x * blockDim.x + threadIdx.x) >> 5;
  int lane = threadIdx.x & 31;
  int tiles_n = Nc >> 4;
  int tm = wid / tiles_n, tn = wid - tm * tiles_n;
  if (tm >= (M >> 4)) return;
  v8f acc = {0.f, 0.f, 0.f, 0.f, 0.f, 0.f, 0.f, 0.f};

  // per-lane strided pointers: no per-iteration index arithmetic
  const _Float16* ap = A + ((size_t)tm * 16 + (lane & 15)) * K + ((lane & 16) ? 16 : 0);
  const _Float16* bp = Bp + ((size_t)tn << 9) + lane * 16;
  size_t bstep = (size_t)tiles_n << 9;
  int KT = K >> 5;

  v8h  alo = *(const v8h*)ap, ahi = *(const v8h*)(ap + 8);
  v16h b   = *(const v16h*)bp;
  for (int kt = 0; kt < KT - 1; ++kt) {
    ap += 32; bp += bstep;
    v8h  nlo = *(const v8h*)ap, nhi = *(const v8h*)(ap + 8);   // prefetch next
    v16h nb  = *(const v16h*)bp;
    v16h a = __builtin_shufflevector(alo, ahi, 0, 1, 2, 3, 4, 5, 6, 7,
                                     8, 9, 10, 11, 12, 13, 14, 15);
    acc = WMMA_F16(a, b, acc);
    alo = nlo; ahi = nhi; b = nb;
  }
  {
    v16h a = __builtin_shufflevector(alo, ahi, 0, 1, 2, 3, 4, 5, 6, 7,
                                     8, 9, 10, 11, 12, 13, 14, 15);
    acc = WMMA_F16(a, b, acc);
  }

  int col = lane & 15;
  int hi8 = (lane & 16) ? 8 : 0;
  int n = tn * 16 + col;
#pragma unroll
  for (int r = 0; r < 8; ++r) {
    size_t m = (size_t)tm * 16 + hi8 + r;
    float v = acc[r];
    if (mode == 0) {
      out_h[m * Nc + n] = (_Float16)v;
    } else if (mode == 1) {
      out_f[m * Nc + n] = v + bias[n] + resid[m * Nc + n];
    } else {
      v += bias[n];
      out_h[m * Nc + n] = (_Float16)(v / (1.f + __expf(-v)));
    }
  }
}

// ---------------- hashing ----------------------------------------------------
__global__ void hash_kernel(const _Float16* __restrict__ qf, const _Float16* __restrict__ kf,
                            const float* __restrict__ coords, const float* __restrict__ sqrtw,
                            const float* __restrict__ alpha,
                            float* __restrict__ qh, float* __restrict__ kh,
                            float* __restrict__ qn2, float* __restrict__ kn2) {
  int gid = blockIdx.x * blockDim.x + threadIdx.x;
  if (gid >= NHEAD * NPTS) return;
  int h = gid >> 15, n = gid & (NPTS - 1);
  const v8h* qr = (const v8h*)(qf + (size_t)n * 512 + h * DH);
  const v8h* kr = (const v8h*)(kf + (size_t)n * 512 + h * DH);
  float qh0 = 0, qh1 = 0, kh0 = 0, kh1 = 0, qs = 0, ks = 0;
#pragma unroll
  for (int c = 0; c < 8; ++c) {
    v8h qv = qr[c], kv = kr[c];
#pragma unroll
    for (int j = 0; j < 8; ++j) {
      int d = c * 8 + j;
      float qd = (float)qv[j], kd = (float)kv[j];
      float a0 = alpha[(h * 67 + d) * 2], a1 = alpha[(h * 67 + d) * 2 + 1];
      qh0 += qd * a0; qh1 += qd * a1;
      kh0 += kd * a0; kh1 += kd * a1;
      qs += qd * qd;  ks += kd * kd;
    }
  }
#pragma unroll
  for (int r = 0; r < 3; ++r) {
    float c = sqrtw[h * 3 + r] * coords[n * 3 + r];
    float a0 = alpha[(h * 67 + 64 + r) * 2], a1 = alpha[(h * 67 + 64 + r) * 2 + 1];
    qh0 += c * a0; qh1 += c * a1;
    kh0 += c * a0; kh1 += c * a1;
    qs += c * c;   ks += c * c;
  }
  qh[(size_t)h * NPTS + n] = qh0;          qh[(size_t)(8 + h) * NPTS + n] = qh1;
  kh[(size_t)h * NPTS + n] = kh0;          kh[(size_t)(8 + h) * NPTS + n] = kh1;
  qn2[(size_t)h * NPTS + n] = 0.5f * qs;   kn2[(size_t)h * NPTS + n] = 0.5f * ks;
}

__global__ void minmax_kernel(const float* __restrict__ qh, const float* __restrict__ kh,
                              float* __restrict__ shift) {
  int lh = blockIdx.x, tid = threadIdx.x;
  const float* q = qh + (size_t)lh * NPTS;
  const float* k = kh + (size_t)lh * NPTS;
  float mx = -1e30f, mn = 1e30f;
  for (int i = tid; i < NPTS; i += 256) {
    float a = q[i], b = k[i];
    mx = fmaxf(mx, fmaxf(a, b));
    mn = fminf(mn, fminf(a, b));
  }
  __shared__ float smx[256], smn[256];
  smx[tid] = mx; smn[tid] = mn;
  __syncthreads();
  for (int s = 128; s; s >>= 1) {
    if (tid < s) { smx[tid] = fmaxf(smx[tid], smx[tid + s]); smn[tid] = fminf(smn[tid], smn[tid + s]); }
    __syncthreads();
  }
  if (tid == 0) shift[lh] = smx[0] - smn[0];
}

__global__ void key_kernel(const float* __restrict__ qh, const float* __restrict__ kh,
                           const int* __restrict__ cshift, const float* __restrict__ shift,
                           float* __restrict__ qkey, int* __restrict__ qidx,
                           float* __restrict__ kkey, int* __restrict__ kidx) {
  int gid = blockIdx.x * blockDim.x + threadIdx.x;
  if (gid >= LHN * NPTS) return;
  int lh = gid >> 15, n = gid & (NPTS - 1);
  float cs = (float)cshift[n] * shift[lh];
  qkey[gid] = qh[gid] + cs; qidx[gid] = n;
  kkey[gid] = kh[gid] + cs; kidx[gid] = n;
}

// bitonic compare-exchange step; blockIdx.y selects q-side vs k-side arrays
__global__ void bitonic_step(float* __restrict__ qkey, int* __restrict__ qidx,
                             float* __restrict__ kkey, int* __restrict__ kidx, int j, int k) {
  int gid = blockIdx.x * blockDim.x + threadIdx.x;
  if (gid >= LHN * NPTS) return;
  float* key = blockIdx.y ? kkey : qkey;
  int*   idx = blockIdx.y ? kidx : qidx;
  int loc = gid & (NPTS - 1);
  int ixj = loc ^ j;
  if (ixj <= loc) return;
  int base = gid - loc;
  float a = key[base + loc], b = key[base + ixj];
  bool up = ((loc & k) == 0);
  if ((a > b) == up) {
    key[base + loc] = b; key[base + ixj] = a;
    int t = idx[base + loc]; idx[base + loc] = idx[base + ixj]; idx[base + ixj] = t;
  }
}

__global__ void invert_kernel(const int* __restrict__ qidx, int* __restrict__ invq) {
  int gid = blockIdx.x * blockDim.x + threadIdx.x;
  if (gid >= LHN * NPTS) return;
  int loc = gid & (NPTS - 1);
  invq[(gid - loc) + qidx[gid]] = loc;
}

// gather q/k/v rows into sorted bucket order; pack scaled coords + 0.5|.|^2
__global__ void gather_kernel(const _Float16* __restrict__ qf, const _Float16* __restrict__ kf,
                              const _Float16* __restrict__ vf,
                              const float* __restrict__ coords, const float* __restrict__ sqrtw,
                              const float* __restrict__ qn2, const float* __restrict__ kn2,
                              const int* __restrict__ qidx, const int* __restrict__ kidx,
                              _Float16* __restrict__ sq, _Float16* __restrict__ sk,
                              _Float16* __restrict__ sv,
                              float4* __restrict__ sqc, float4* __restrict__ skc) {
  int gid = blockIdx.x * blockDim.x + threadIdx.x;
  if (gid >= LHN * NPTS) return;
  int lh = gid >> 15, h = lh & 7;
  int srcq = qidx[gid], srck = kidx[gid];
  const uint4* gq = (const uint4*)(qf + (size_t)srcq * 512 + h * DH);
  const uint4* gk = (const uint4*)(kf + (size_t)srck * 512 + h * DH);
  const uint4* gv = (const uint4*)(vf + (size_t)srck * 512 + h * DH);
  uint4* dq = (uint4*)(sq + (size_t)gid * DH);
  uint4* dk = (uint4*)(sk + (size_t)gid * DH);
  uint4* dv = (uint4*)(sv + (size_t)gid * DH);
#pragma unroll
  for (int j = 0; j < 8; ++j) { dq[j] = gq[j]; dk[j] = gk[j]; dv[j] = gv[j]; }
  float4 qc, kc;
  qc.x = sqrtw[h * 3 + 0] * coords[srcq * 3 + 0];
  qc.y = sqrtw[h * 3 + 1] * coords[srcq * 3 + 1];
  qc.z = sqrtw[h * 3 + 2] * coords[srcq * 3 + 2];
  qc.w = qn2[(size_t)h * NPTS + srcq];
  kc.x = sqrtw[h * 3 + 0] * coords[srck * 3 + 0];
  kc.y = sqrtw[h * 3 + 1] * coords[srck * 3 + 1];
  kc.z = sqrtw[h * 3 + 2] * coords[srck * 3 + 2];
  kc.w = kn2[(size_t)h * NPTS + srck];
  sqc[gid] = qc; skc[gid] = kc;
}

// ---------------- per-bucket RBF attention (WMMA) ----------------------------
// one 256-thread block (8 waves) per (l,h,bucket); K and kc staged via
// CDNA5 async global->LDS DMA; V transposed through VGPRs; P staged in LDS.
__global__ void bucket_attn(const _Float16* __restrict__ sq, const _Float16* __restrict__ sk,
                            const _Float16* __restrict__ sv,
                            const float4* __restrict__ sqc, const float4* __restrict__ skc,
                            float* __restrict__ o, float* __restrict__ denom) {
  extern __shared__ char smem[];
  _Float16* lk  = (_Float16*)smem;             // 128x64 f16 (row-major, key x d)
  _Float16* lvT = lk + 128 * 64;               // 64x128 f16 (d x key, transposed)
  float4*   lkc = (float4*)(lvT + 64 * 128);   // 128 float4
  _Float16* lp  = (_Float16*)(lkc + 128);      // 128x128 f16 exp-scores

  int lh = blockIdx.x >> 8;
  int bucket = blockIdx.x & (NBUCK - 1);
  size_t rowbase = (size_t)lh * NPTS + bucket * 128;
  int tid = threadIdx.x;

  // async DMA: K tile (16KB) and kc vectors (2KB) straight into LDS
  {
    const uint4* gk = (const uint4*)(sk + rowbase * DH);
    unsigned lk_off  = (unsigned)(uintptr_t)lk;
    unsigned lkc_off = (unsigned)(uintptr_t)lkc;
    for (int i = tid; i < 1024; i += 256)
      async_load_b128(lk_off + i * 16, gk + i);
    if (tid < 128)
      async_load_b128(lkc_off + tid * 16, skc + rowbase + tid);
    // V transpose: vector read through VGPRs, scattered LDS writes
    const v8h* gv = (const v8h*)(sv + rowbase * DH);
    for (int i = tid; i < 1024; i += 256) {
      int j = i >> 3, c = i & 7;          // key row j, group of 8 d's
      v8h v = gv[i];
#pragma unroll
      for (int t = 0; t < 8; ++t) lvT[(c * 8 + t) * 128 + j] = v[t];
    }
    wait_asynccnt0();
  }
  __syncthreads();

  int w = tid >> 5, lane = tid & 31;
  int col = lane & 15, hi8 = (lane & 16) ? 8 : 0;

  const _Float16* Aq = sq + (rowbase + w * 16) * DH;
  v16h a0 = load_a_frag(Aq, DH, lane);
  v16h a1 = load_a_frag(Aq + 32, DH, lane);
  float4 qc[8];
#pragma unroll
  for (int r = 0; r < 8; ++r) qc[r] = sqc[rowbase + w * 16 + hi8 + r];

  // scores: exp(min(q.k + qc.kc - qn2 - kn2, 0)) -> LDS P (own 16x128 slab)
  for (int jt = 0; jt < 8; ++jt) {
    v8f acc = {0.f, 0.f, 0.f, 0.f, 0.f, 0.f, 0.f, 0.f};
    v16h b0 = load_bT_frag(lk + jt * 16 * DH, DH, lane);
    v16h b1 = load_bT_frag(lk + jt * 16 * DH + 32, DH, lane);
    acc = WMMA_F16(a0, b0, acc);
    acc = WMMA_F16(a1, b1, acc);
    float4 kc = lkc[jt * 16 + col];
#pragma unroll
    for (int r = 0; r < 8; ++r) {
      float s = acc[r] + qc[r].x * kc.x + qc[r].y * kc.y + qc[r].z * kc.z - qc[r].w - kc.w;
      lp[(w * 16 + hi8 + r) * 128 + jt * 16 + col] = (_Float16)__expf(fminf(s, 0.f));
    }
  }
  __syncthreads();

  // per-row denominators (each lane sums half a row, vectorized LDS reads)
  {
    int rloc = w * 16 + (lane >> 1);
    const v8h* prow = (const v8h*)(lp + rloc * 128 + (lane & 1) * 64);
    float s = 0.f;
#pragma unroll
    for (int c = 0; c < 8; ++c) {
      v8h v = prow[c];
#pragma unroll
      for (int t = 0; t < 8; ++t) s += (float)v[t];
    }
    s += __shfl_xor(s, 1);
    if ((lane & 1) == 0) denom[rowbase + rloc] = s + 1e-20f;
  }

  // P @ V (B-fragments contiguous from transposed V)
  for (int ct = 0; ct < 4; ++ct) {
    v8f acc = {0.f, 0.f, 0.f, 0.f, 0.f, 0.f, 0.f, 0.f};
#pragma unroll
    for (int kk = 0; kk < 4; ++kk) {
      v16h a = load_a_frag(lp + (w * 16) * 128 + kk * 32, 128, lane);
      v16h b = load_bT_frag(lvT + (ct * 16) * 128 + kk * 32, 128, lane);
      acc = WMMA_F16(a, b, acc);
    }
#pragma unroll
    for (int r = 0; r < 8; ++r)
      o[(rowbase + w * 16 + hi8 + r) * DH + ct * 16 + col] = acc[r];
  }
}

// unsort via inverse q-permutation, combine over L, normalize, emit f16 (N,512)
__global__ void combine_kernel(const float* __restrict__ o, const float* __restrict__ den,
                               const int* __restrict__ invq, _Float16* __restrict__ attn) {
  int gid = blockIdx.x * blockDim.x + threadIdx.x;   // N*H*4 chunks of 16
  if (gid >= NPTS * NHEAD * 4) return;
  int c = gid & 3, h = (gid >> 2) & 7, n = gid >> 5;
  float acc[16];
#pragma unroll
  for (int j = 0; j < 16; ++j) acc[j] = 0.f;
  float dsum = 0.f;
#pragma unroll
  for (int l = 0; l < 2; ++l) {
    int lh = l * 8 + h;
    int i = invq[(size_t)lh * NPTS + n];
    dsum += den[(size_t)lh * NPTS + i];
    const float4* orow = (const float4*)(o + ((size_t)lh * NPTS + i) * DH + c * 16);
#pragma unroll
    for (int j = 0; j < 4; ++j) {
      float4 v = orow[j];
      acc[4 * j] += v.x; acc[4 * j + 1] += v.y; acc[4 * j + 2] += v.z; acc[4 * j + 3] += v.w;
    }
  }
  float inv = 1.f / dsum;
  _Float16* dst = attn + (size_t)n * 512 + h * DH + c * 16;
#pragma unroll
  for (int j = 0; j < 16; ++j) dst[j] = (_Float16)(acc[j] * inv);
}

// ---------------- host driver ------------------------------------------------

extern "C" void kernel_launch(void* const* d_in, const int* in_sizes, int n_in,
                              void* d_out, int out_size, void* d_ws, size_t ws_size,
                              hipStream_t stream) {
  const float* x      = (const float*)d_in[0];
  const float* coords = (const float*)d_in[1];
  const int*   cshift = (const int*)d_in[2];
  const float* wq     = (const float*)d_in[3];
  const float* wk     = (const float*)d_in[4];
  const float* wv     = (const float*)d_in[5];
  const float* out_w  = (const float*)d_in[6];
  const float* out_b  = (const float*)d_in[7];
  const float* n1g    = (const float*)d_in[8];
  const float* n1b    = (const float*)d_in[9];
  const float* n2g    = (const float*)d_in[10];
  const float* n2b    = (const float*)d_in[11];
  const float* ff1w   = (const float*)d_in[12];
  const float* ff1b   = (const float*)d_in[13];
  const float* ff2w   = (const float*)d_in[14];
  const float* ff2b   = (const float*)d_in[15];
  const float* w_rpe  = (const float*)d_in[16];
  const float* alpha  = (const float*)d_in[17];
  float* outp = (float*)d_out;

  uintptr_t p = (uintptr_t)d_ws;
  auto alloc = [&](size_t bytes) -> void* {
    void* r = (void*)p;
    p += (bytes + 255) & ~(size_t)255;
    return r;
  };
  float*     sqrtw = (float*)alloc(24 * 4);
  float*     shift = (float*)alloc(LHN * 4);
  _Float16*  xn    = (_Float16*)alloc((size_t)NPTS * 64 * 2);
  _Float16*  wq_p  = (_Float16*)alloc(64 * 512 * 2);
  _Float16*  wk_p  = (_Float16*)alloc(64 * 512 * 2);
  _Float16*  wv_p  = (_Float16*)alloc(64 * 512 * 2);
  _Float16*  ow_p  = (_Float16*)alloc(512 * 64 * 2);
  _Float16*  f1_p  = (_Float16*)alloc(64 * 64 * 2);
  _Float16*  f2_p  = (_Float16*)alloc(64 * 64 * 2);
  _Float16*  qf    = (_Float16*)alloc((size_t)NPTS * 512 * 2);
  _Float16*  kf    = (_Float16*)alloc((size_t)NPTS * 512 * 2);
  _Float16*  vf    = (_Float16*)alloc((size_t)NPTS * 512 * 2);
  float*     qhash = (float*)alloc((size_t)LHN * NPTS * 4);
  float*     khash = (float*)alloc((size_t)LHN * NPTS * 4);
  float*     qn2   = (float*)alloc((size_t)NHEAD * NPTS * 4);
  float*     kn2   = (float*)alloc((size_t)NHEAD * NPTS * 4);
  float*     qkey  = (float*)alloc((size_t)LHN * NPTS * 4);
  float*     kkey  = (float*)alloc((size_t)LHN * NPTS * 4);
  int*       qidx  = (int*)alloc((size_t)LHN * NPTS * 4);
  int*       kidx  = (int*)alloc((size_t)LHN * NPTS * 4);
  int*       invq  = (int*)alloc((size_t)LHN * NPTS * 4);
  _Float16*  sq    = (_Float16*)alloc((size_t)LHN * NPTS * 64 * 2);
  _Float16*  sk    = (_Float16*)alloc((size_t)LHN * NPTS * 64 * 2);
  _Float16*  sv    = (_Float16*)alloc((size_t)LHN * NPTS * 64 * 2);
  float4*    sqc   = (float4*)alloc((size_t)LHN * NPTS * 16);
  float4*    skc   = (float4*)alloc((size_t)LHN * NPTS * 16);
  float*     obuf  = (float*)alloc((size_t)LHN * NPTS * 64 * 4);
  float*     den   = (float*)alloc((size_t)LHN * NPTS * 4);
  _Float16*  attn  = (_Float16*)alloc((size_t)NPTS * 512 * 2);
  float*     x2    = (float*)alloc((size_t)NPTS * 64 * 4);
  _Float16*  h16   = (_Float16*)alloc((size_t)NPTS * 64 * 2);
  _Float16*  sil   = (_Float16*)alloc((size_t)NPTS * 64 * 2);

  // 0) RPE scales + weight packing (WMMA fragment order)
  rpe_kernel<<<1, 32, 0, stream>>>(w_rpe, sqrtw);
  pack_weight_kernel<<<(64 * 512 + 255) / 256, 256, 0, stream>>>(wq, wq_p, 64, 512);
  pack_weight_kernel<<<(64 * 512 + 255) / 256, 256, 0, stream>>>(wk, wk_p, 64, 512);
  pack_weight_kernel<<<(64 * 512 + 255) / 256, 256, 0, stream>>>(wv, wv_p, 64, 512);
  pack_weight_kernel<<<(512 * 64 + 255) / 256, 256, 0, stream>>>(out_w, ow_p, 512, 64);
  pack_weight_kernel<<<(64 * 64 + 255) / 256, 256, 0, stream>>>(ff1w, f1_p, 64, 64);
  pack_weight_kernel<<<(64 * 64 + 255) / 256, 256, 0, stream>>>(ff2w, f2_p, 64, 64);

  // 1) LN1 -> f16
  ln_kernel<<<NPTS * 32 / 256, 256, 0, stream>>>(x, n1g, n1b, xn, NPTS);

  // 2) QKV projections: (N,64)@(64,512) WMMA
  {
    int blocks = (NPTS / 16) * (512 / 16) / 8;
    gemm_wmma<<<blocks, 256, 0, stream>>>(xn, wq_p, NPTS, 64, 512, nullptr, nullptr, qf, nullptr, 0);
    gemm_wmma<<<blocks, 256, 0, stream>>>(xn, wk_p, NPTS, 64, 512, nullptr, nullptr, kf, nullptr, 0);
    gemm_wmma<<<blocks, 256, 0, stream>>>(xn, wv_p, NPTS, 64, 512, nullptr, nullptr, vf, nullptr, 0);
  }

  // 3) E2LSH hashes, norms, shift range, sort keys
  hash_kernel<<<NHEAD * NPTS / 256, 256, 0, stream>>>(qf, kf, coords, sqrtw, alpha,
                                                      qhash, khash, qn2, kn2);
  minmax_kernel<<<LHN, 256, 0, stream>>>(qhash, khash, shift);
  key_kernel<<<LHN * NPTS / 256, 256, 0, stream>>>(qhash, khash, cshift, shift,
                                                   qkey, qidx, kkey, kidx);

  // 4) bitonic argsort of 16 segments of 32768 (q and k sides via grid.y)
  for (int k = 2; k <= NPTS; k <<= 1)
    for (int j = k >> 1; j > 0; j >>= 1)
      bitonic_step<<<dim3(LHN * NPTS / 256, 2), 256, 0, stream>>>(qkey, qidx, kkey, kidx, j, k);

  invert_kernel<<<LHN * NPTS / 256, 256, 0, stream>>>(qidx, invq);

  // 5) gather into bucket order
  gather_kernel<<<LHN * NPTS / 256, 256, 0, stream>>>(qf, kf, vf, coords, sqrtw, qn2, kn2,
                                                      qidx, kidx, sq, sk, sv, sqc, skc);

  // 6) per-bucket RBF attention (WMMA, async-LDS staged)
  bucket_attn<<<LHN * NBUCK, 256, 66 * 1024 + 2048, stream>>>(sq, sk, sv, sqc, skc, obuf, den);

  // 7) unsort + combine over L
  combine_kernel<<<NPTS * NHEAD * 4 / 256, 256, 0, stream>>>(obuf, den, invq, attn);

  // 8) output projection + residual: x2 = x + attn@out_w + b
  gemm_wmma<<<(NPTS / 16) * (64 / 16) / 8, 256, 0, stream>>>(attn, ow_p, NPTS, 512, 64,
                                                             out_b, x, nullptr, x2, 1);
  // 9) FFN
  ln_kernel<<<NPTS * 32 / 256, 256, 0, stream>>>(x2, n2g, n2b, h16, NPTS);
  gemm_wmma<<<(NPTS / 16) * (64 / 16) / 8, 256, 0, stream>>>(h16, f1_p, NPTS, 64, 64,
                                                             ff1b, nullptr, sil, nullptr, 2);
  gemm_wmma<<<(NPTS / 16) * (64 / 16) / 8, 256, 0, stream>>>(sil, f2_p, NPTS, 64, 64,
                                                             ff2b, x2, nullptr, outp, 1);
}
